// LSTMCell_13056700580135
// MI455X (gfx1250) — matrix operand order repaired and checked by reading, not compiled, not run on hardware
//
#include <hip/hip_runtime.h>
#include <stdint.h>

typedef __attribute__((ext_vector_type(16))) __bf16 v16bf;
typedef __attribute__((ext_vector_type(8)))  float  v8f;

#define B_ROWS 262144
#define HDIM   128
#define KDIM   256            // I + H concatenated along K
#define NGATES 4
#define WAVES  8
#define THREADS (WAVES * 32)
#define MT     16             // 16-row M tiles per workgroup
#define ROWS_PER_WG (MT * 16)

union BFrag { uint4 u4[2]; v16bf v; };

__device__ __forceinline__ uint32_t f32_to_bf16_bits(float f) {
  union { float f; uint32_t u; } x; x.f = f;
  return (x.u + 0x7FFFu + ((x.u >> 16) & 1u)) >> 16;   // RNE
}

__device__ __forceinline__ float fast_sigmoid(float x) {
  return 1.0f / (1.0f + __expf(-x));
}
__device__ __forceinline__ float fast_tanh(float x) {
  float e2 = __expf(-2.0f * x);
  return (1.0f - e2) / (1.0f + e2);
}

// ---- pack 8 fp32 weight mats -> gate-major bf16 W[512][256] in workspace ----
__global__ __launch_bounds__(256) void pack_weights(
    const float* __restrict__ Wxi, const float* __restrict__ Whi,
    const float* __restrict__ Wxf, const float* __restrict__ Whf,
    const float* __restrict__ Wxo, const float* __restrict__ Who,
    const float* __restrict__ Wxg, const float* __restrict__ Whg,
    uint16_t* __restrict__ wb) {
  int idx = blockIdx.x * 256 + threadIdx.x;    // 0..131071
  int n    = idx >> 8;                         // 0..511 gate-major output row
  int k    = idx & 255;                        // 0..255 (input 0-127 | hidden 128-255)
  int gate = n >> 7;
  int row  = n & 127;
  const float* src;
  if (k < 128) src = (gate == 0) ? Wxi : (gate == 1) ? Wxf : (gate == 2) ? Wxo : Wxg;
  else         src = (gate == 0) ? Whi : (gate == 1) ? Whf : (gate == 2) ? Who : Whg;
  wb[idx] = (uint16_t)f32_to_bf16_bits(src[row * 128 + (k & 127)]);
}

// ---- fused LSTM cell: bf16 WMMA GEMM + in-register activations ----
__global__ __launch_bounds__(THREADS, 1) void lstm_wmma(
    const float* __restrict__ inputs, const float* __restrict__ h_prev,
    const float* __restrict__ c_prev, const uint16_t* __restrict__ wb,
    float* __restrict__ out_h, float* __restrict__ out_c) {
  // double-buffered A tile, laid out in WMMA A-fragment order
  __shared__ uint4 aTile4[2][512];                     // 2 x 8 KB

  const int tid  = threadIdx.x;
  const int lane = tid & 31;
  const int wave = tid >> 5;
  const int half = lane >> 4;       // 0: lanes 0-15, 1: lanes 16-31
  const int l16  = lane & 15;
  const int wcol = wave * 16 + l16; // H-column owned by this lane (per gate)

  // --- preload B fragments: weights W[n][k], lane holds B column = W row ---
  // B 32x16 fragment in contiguous k order used here (A in LDS matches): lane n -> W[n][kbase+0..15], lane n+16 -> W[n][kbase+16..31]
  BFrag bf[NGATES][8];
#pragma unroll
  for (int g = 0; g < NGATES; ++g) {
    const int n = g * HDIM + wcol;
#pragma unroll
    for (int ks = 0; ks < 8; ++ks) {
      const uint4* p = (const uint4*)(wb + (size_t)n * KDIM + ks * 32 + half * 16);
      bf[g][ks].u4[0] = p[0];
      bf[g][ks].u4[1] = p[1];
    }
  }

  const long m_base = (long)blockIdx.x * ROWS_PER_WG;
  const float2* in2 = (const float2*)inputs;
  const float2* hp2 = (const float2*)h_prev;

  // per-thread staging of one A tile (8 float2 = 16 fp32)
  float2 stage[8];

  // destination LDS dword offsets for this thread's 8 bf16-pairs (tile-invariant)
  int dwoff[8];
#pragma unroll
  for (int i = 0; i < 8; ++i) {
    int idx = i * THREADS + tid;       // 0..2047 (row, k-pair)
    int row = idx >> 7;                // 0..15
    int k2  = idx & 127;               // pair index, k = 2*k2
    int k   = k2 * 2;
    int ks  = k >> 5;
    int r5  = k & 31;
    int c   = (r5 >> 3) & 1;           // chunk (VGPR 0-3 vs 4-7)
    int hf  = (r5 >> 4) & 1;           // lane half
    int pos = r5 & 7;                  // element within 16B chunk
    dwoff[i] = (((ks * 2 + c) * 32) + (hf * 16 + row)) * 4 + (pos >> 1);
  }

  // --- global load of one 16x256 A tile into registers ---
  auto loadA = [&](long m0, float2* s) {
#pragma unroll
    for (int i = 0; i < 8; ++i) {
      int idx = i * THREADS + tid;
      int row = idx >> 7;
      int k2  = idx & 127;
      s[i] = (k2 < 64) ? in2[(m0 + row) * 64 + k2]
                       : hp2[(m0 + row) * 64 + (k2 - 64)];
    }
  };
  // --- convert + scatter staged tile into LDS fragment layout ---
  auto storeA = [&](const float2* s, int buf) {
    uint32_t* t32 = (uint32_t*)aTile4[buf];
#pragma unroll
    for (int i = 0; i < 8; ++i) {
      uint32_t packed = f32_to_bf16_bits(s[i].x) | (f32_to_bf16_bits(s[i].y) << 16);
      t32[dwoff[i]] = packed;
    }
  };

  // prologue: tile 0 -> LDS buffer 0
  loadA(m_base, stage);
  storeA(stage, 0);

  for (int t = 0; t < MT; ++t) {
    const long m0  = m_base + (long)t * 16;
    const int  buf = t & 1;
    __syncthreads();   // tile t resident in aTile4[buf]

    // issue next tile's global loads early; they drain during the WMMAs
    if (t + 1 < MT) loadA(m0 + 16, stage);

    const uint4* tile = aTile4[buf];
    v8f acc0 = {}, acc1 = {}, acc2 = {}, acc3 = {};

    // software-pipelined A fragments: load ks+1 before issuing WMMAs for ks
    BFrag a;
    a.u4[0] = tile[0 * 32 + lane];
    a.u4[1] = tile[1 * 32 + lane];
#pragma unroll
    for (int ks = 0; ks < 8; ++ks) {
      BFrag an;
      if (ks < 7) {
        an.u4[0] = tile[((ks + 1) * 2 + 0) * 32 + lane];
        an.u4[1] = tile[((ks + 1) * 2 + 1) * 32 + lane];
      }
      acc0 = __builtin_amdgcn_wmma_f32_16x16x32_bf16(false, a.v, false, bf[0][ks].v,
                                                     (short)0, acc0, false, false);
      acc1 = __builtin_amdgcn_wmma_f32_16x16x32_bf16(false, a.v, false, bf[1][ks].v,
                                                     (short)0, acc1, false, false);
      acc2 = __builtin_amdgcn_wmma_f32_16x16x32_bf16(false, a.v, false, bf[2][ks].v,
                                                     (short)0, acc2, false, false);
      acc3 = __builtin_amdgcn_wmma_f32_16x16x32_bf16(false, a.v, false, bf[3][ks].v,
                                                     (short)0, acc3, false, false);
      if (ks < 7) a = an;
    }

    // convert + scatter next tile into the other LDS buffer (overlaps epilogue)
    if (t + 1 < MT) storeA(stage, buf ^ 1);

    // --- C/D layout: VGPR r -> M = r + 8*half, N = lane&15 ---
#pragma unroll
    for (int r = 0; r < 8; ++r) {
      const long m   = m0 + r + half * 8;
      const float cp = c_prev[m * HDIM + wcol];
      float it = fast_sigmoid(acc0[r]);
      float ft = fast_sigmoid(acc1[r]);
      float ot = fast_sigmoid(acc2[r]);
      float gt = fast_tanh(acc3[r]);
      float ct = ft * cp + it * gt;
      float ht = ot * fast_tanh(ct);
      out_h[m * HDIM + wcol] = ht;
      out_c[m * HDIM + wcol] = ct;
    }
  }
}

extern "C" void kernel_launch(void* const* d_in, const int* in_sizes, int n_in,
                              void* d_out, int out_size, void* d_ws, size_t ws_size,
                              hipStream_t stream) {
  const float* inputs = (const float*)d_in[0];
  const float* h_prev = (const float*)d_in[1];
  const float* c_prev = (const float*)d_in[2];
  const float* W_ii = (const float*)d_in[3];
  const float* W_hi = (const float*)d_in[4];
  const float* W_if = (const float*)d_in[5];
  const float* W_hf = (const float*)d_in[6];
  const float* W_io = (const float*)d_in[7];
  const float* W_ho = (const float*)d_in[8];
  const float* W_ig = (const float*)d_in[9];
  const float* W_hg = (const float*)d_in[10];

  uint16_t* wb = (uint16_t*)d_ws;                 // 512*256 bf16 = 256 KB scratch
  float* out_h = (float*)d_out;                   // [B, H]
  float* out_c = out_h + (long)B_ROWS * HDIM;     // [B, H]

  pack_weights<<<512, 256, 0, stream>>>(W_ii, W_hi, W_if, W_hf,
                                        W_io, W_ho, W_ig, W_hg, wb);
  lstm_wmma<<<B_ROWS / ROWS_PER_WG, THREADS, 0, stream>>>(
      inputs, h_prev, c_prev, wb, out_h, out_c);
}
